// Attention_64312840290684
// MI455X (gfx1250) — matrix operand order repaired and checked by reading, not compiled, not run on hardware
//
#include <hip/hip_runtime.h>
#include <hip/hip_bf16.h>

// ---------------------------------------------------------------------------
// MHA block for MI455X (gfx1250): bf16 WMMA everywhere, fp32 accumulate,
// async global->LDS DMA (ASYNCcnt) for tile staging.
//   B=8, N=1024, C=768, H=12, hd=64, SCALE=0.125
// ---------------------------------------------------------------------------

typedef __bf16 bf16;
typedef __attribute__((ext_vector_type(16))) __bf16 v16bf;
typedef __attribute__((ext_vector_type(8)))  __bf16 v8bf;
typedef __attribute__((ext_vector_type(8)))  float  v8f;

#define EMBED 768
#define HEADS 12
#define HD    64
#define BATCH 8
#define SEQ   1024
#define TOKENS (BATCH * SEQ)     // 8192
#define QKV_DIM (3 * EMBED)      // 2304
#define ATTN_SCALE 0.125f

__device__ __forceinline__ bf16 f2bf(float x) { return (bf16)x; }

__device__ __forceinline__ v8f wmma_bf16(v16bf a, v16bf b, v8f c) {
  // D = A(16x32 bf16) * B(32x16 bf16) + C(16x16 f32)
  return __builtin_amdgcn_wmma_f32_16x16x32_bf16(
      /*neg_a=*/false, a, /*neg_b=*/false, b,
      /*c_mod=*/(short)0, c, /*reuse_a=*/false, /*reuse_b=*/false);
}

// Async DMA: 16 bytes global -> LDS per lane, tracked by ASYNCcnt.
__device__ __forceinline__ void async_ld_b128(unsigned lds_addr, const void* gaddr) {
  asm volatile("global_load_async_to_lds_b128 %0, %1, off"
               :: "v"(lds_addr), "v"(gaddr) : "memory");
}
__device__ __forceinline__ void wait_async0() {
  asm volatile("s_wait_asynccnt 0" ::: "memory");
}
__device__ __forceinline__ void wait_async4() {
  asm volatile("s_wait_asynccnt 4" ::: "memory");
}
__device__ __forceinline__ unsigned lds_off(const void* p) {
  return (unsigned)(size_t)p;   // low 32 bits of an LDS generic address
}

// A fragment (16x32, row-major tile in LDS, leading dim `ld` elements).
// ISA layout: lanes 0-15 hold row M=lane with K={0..7,16..23};
//             lanes 16-31 hold row M=lane-16 with K={8..15,24..31}.
__device__ __forceinline__ v16bf load_frag_a(const bf16* lds, int ld, int row0, int k0) {
  int lane = threadIdx.x & 31;
  int r    = row0 + (lane & 15);
  int kh   = (lane >> 4) * 8;            // 0 or 8
  const v8bf* p0 = (const v8bf*)(lds + r * ld + k0 + kh);
  const v8bf* p1 = (const v8bf*)(lds + r * ld + k0 + kh + 16);
  v8bf lo = *p0, hi = *p1;
  v16bf f;
#pragma unroll
  for (int i = 0; i < 8; ++i) { f[i] = lo[i]; f[i + 8] = hi[i]; }
  return f;
}

// B fragment (32x16). LDS tile stores each output column n as a contiguous
// K-run (i.e. tile[n][k], leading dim `ld`).  ISA layout: lanes 0-15 hold
// column N=lane with K={0..7,16..23}, lanes 16-31 hold K={8..15,24..31}.
__device__ __forceinline__ v16bf load_frag_b(const bf16* lds, int ld, int k0, int n0) {
  int lane = threadIdx.x & 31;
  int n    = n0 + (lane & 15);
  int kh   = (lane >> 4) * 8;            // 0 or 8
  const v8bf* p0 = (const v8bf*)(lds + n * ld + k0 + kh);
  const v8bf* p1 = (const v8bf*)(lds + n * ld + k0 + kh + 16);
  v8bf lo = *p0, hi = *p1;
  v16bf f;
#pragma unroll
  for (int i = 0; i < 8; ++i) { f[i] = lo[i]; f[i + 8] = hi[i]; }
  return f;
}

// ---------------------------------------------------------------------------
// fp32 -> bf16 conversion (one-time repack of x and weights)
// ---------------------------------------------------------------------------
__global__ void __launch_bounds__(256) cvt_f32_bf16(const float* __restrict__ in,
                                                    bf16* __restrict__ out, int n) {
  for (int i = blockIdx.x * 256 + threadIdx.x; i < n; i += gridDim.x * 256)
    out[i] = f2bf(in[i]);
}

// ---------------------------------------------------------------------------
// Tiled bf16 GEMM:  out[m,d] = sum_k A[m,k] * W[d,k] + bias[d]
//   A: MxK row-major bf16, W: NoutxK row-major bf16 (i.e. we compute A*W^T).
//   mode 0: write fp32 to outf (output projection)
//   mode 1: scatter bf16 to Q (scaled), K, V in [B,H,N,hd] layout (QKV proj)
// 64x64 block tile, 128 threads (4 waves), each wave 32x32 = 2x2 WMMA frags.
// Double-buffered async global->LDS staging: DMA for tile k+1 overlaps the
// WMMAs on tile k; s_wait_asynccnt<=4 (in-order per wave) gates the buffer.
// ---------------------------------------------------------------------------
__global__ void __launch_bounds__(128) gemm_bf16_kernel(
    const bf16* __restrict__ A, const bf16* __restrict__ W,
    const float* __restrict__ bias, float* __restrict__ outf,
    bf16* __restrict__ Qb, bf16* __restrict__ Kb, bf16* __restrict__ Vb,
    int M, int K, int Nout, int mode) {
  __shared__ __align__(16) bf16 As[2][64 * 32];   // [buf][row][k]
  __shared__ __align__(16) bf16 Ws[2][64 * 32];   // [buf][n][k]

  const int mtile = blockIdx.x * 64;
  const int ntile = blockIdx.y * 64;
  const int tid  = threadIdx.x;
  const int wave = tid >> 5;
  const int lane = tid & 31;
  const int half = lane >> 4;
  const int nc   = lane & 15;
  const int msub = (wave & 1) * 32;
  const int nsub = (wave >> 1) * 32;

  v8f acc[2][2] = {};

  const int lrow = tid >> 1;
  const int lc   = (tid & 1) * 16;
  const bf16* Ag = A + (size_t)(mtile + lrow) * K + lc;
  const bf16* Wg = W + (size_t)(ntile + lrow) * K + lc;
  const unsigned ldsA[2] = { lds_off(&As[0][lrow * 32 + lc]),
                             lds_off(&As[1][lrow * 32 + lc]) };
  const unsigned ldsW[2] = { lds_off(&Ws[0][lrow * 32 + lc]),
                             lds_off(&Ws[1][lrow * 32 + lc]) };

  // Prologue: stage k0=0 into buffer 0 (4 x b128 per lane).
  async_ld_b128(ldsA[0],      Ag);
  async_ld_b128(ldsA[0] + 16, Ag + 8);
  async_ld_b128(ldsW[0],      Wg);
  async_ld_b128(ldsW[0] + 16, Wg + 8);

  int buf = 0;
  for (int k0 = 0; k0 < K; k0 += 32, buf ^= 1) {
    __syncthreads();                       // prior readers of buf^1 are done
    const bool more = (k0 + 32 < K);       // uniform
    if (more) {
      const bf16* An = Ag + k0 + 32;
      const bf16* Wn = Wg + k0 + 32;
      async_ld_b128(ldsA[buf ^ 1],      An);
      async_ld_b128(ldsA[buf ^ 1] + 16, An + 8);
      async_ld_b128(ldsW[buf ^ 1],      Wn);
      async_ld_b128(ldsW[buf ^ 1] + 16, Wn + 8);
      wait_async4();                       // current buffer's 4 copies landed
    } else {
      wait_async0();
    }
    __syncthreads();                       // all waves see current buffer

    v16bf a0 = load_frag_a(As[buf], 32, msub,      0);
    v16bf a1 = load_frag_a(As[buf], 32, msub + 16, 0);
    v16bf b0 = load_frag_b(Ws[buf], 32, 0, nsub);
    v16bf b1 = load_frag_b(Ws[buf], 32, 0, nsub + 16);
    acc[0][0] = wmma_bf16(a0, b0, acc[0][0]);
    acc[0][1] = wmma_bf16(a0, b1, acc[0][1]);
    acc[1][0] = wmma_bf16(a1, b0, acc[1][0]);
    acc[1][1] = wmma_bf16(a1, b1, acc[1][1]);
  }

  // Epilogue.  C/D layout: lanes 0-15: M=vgpr, N=lane; lanes 16-31: M=vgpr+8.
#pragma unroll
  for (int i = 0; i < 2; ++i)
#pragma unroll
    for (int j = 0; j < 2; ++j)
#pragma unroll
      for (int r = 0; r < 8; ++r) {
        int m = mtile + msub + i * 16 + half * 8 + r;
        int d = ntile + nsub + j * 16 + nc;
        float val = acc[i][j][r] + bias[d];
        if (mode == 0) {
          outf[(size_t)m * Nout + d] = val;
        } else {
          int which = d / EMBED;
          int rem   = d - which * EMBED;
          int h     = rem >> 6;        // /64
          int e     = rem & 63;
          int btok  = m >> 10;         // /1024
          int ntok  = m & 1023;
          size_t idx = (((size_t)btok * HEADS + h) * SEQ + ntok) * HD + e;
          if (which == 0)      Qb[idx] = f2bf(val * ATTN_SCALE);
          else if (which == 1) Kb[idx] = f2bf(val);
          else                 Vb[idx] = f2bf(val);
        }
      }
}

// ---------------------------------------------------------------------------
// Flash attention: one block per (b*h, 64-query tile).  128 threads = 4 waves;
// wave w owns query rows w*16..w*16+15 across ALL 64 key columns (1x4 frags),
// so online-softmax row reductions stay inside a half-wave (shfl_xor 1/2/4/8).
// Q and K tiles are staged with async global->LDS DMA; V is transposed through
// VGPRs (needed for the [e][kk] B-layout of the P*V GEMM).
// ---------------------------------------------------------------------------
__global__ void __launch_bounds__(128) attn_kernel(
    const bf16* __restrict__ Qb, const bf16* __restrict__ Kb,
    const bf16* __restrict__ Vb, bf16* __restrict__ ctx) {
  __shared__ __align__(16) bf16 Qs [64 * 64];  // [q][d]
  __shared__ __align__(16) bf16 Ks [64 * 64];  // [kk][d]  (B for S = Q K^T)
  __shared__ __align__(16) bf16 Vts[64 * 64];  // [e][kk]  (B for O = P V)
  __shared__ __align__(16) bf16 Ps [64 * 64];  // [q][kk]  (A for O = P V)

  const int bh  = blockIdx.x >> 4;   // 0..95 = b*12+h
  const int qb  = blockIdx.x & 15;
  const int tid = threadIdx.x;
  const int wave = tid >> 5;
  const int lane = tid & 31;
  const int half = lane >> 4;
  const int nc   = lane & 15;

  const int row = tid >> 1;
  const int c0  = (tid & 1) * 32;

  // Stage the 64x64 Q tile via async DMA (Q pre-scaled by 1/sqrt(hd)).
  {
    const bf16* g = Qb + (((size_t)bh * SEQ) + qb * 64) * HD + row * HD + c0;
    unsigned l = lds_off(&Qs[row * 64 + c0]);
    async_ld_b128(l,      g);
    async_ld_b128(l + 16, g + 8);
    async_ld_b128(l + 32, g + 16);
    async_ld_b128(l + 48, g + 24);
  }

  v8f o[4] = {};
  float mst[8], lst[8];
#pragma unroll
  for (int r = 0; r < 8; ++r) { mst[r] = -1e30f; lst[r] = 0.0f; }

  for (int kb = 0; kb < SEQ / 64; ++kb) {
    __syncthreads();   // prev-iter consumers of Ks/Vts are done
    {
      // K tile: straight copy -> async DMA.
      const bf16* kg = Kb + (((size_t)bh * SEQ) + kb * 64) * HD + row * HD + c0;
      unsigned l = lds_off(&Ks[row * 64 + c0]);
      async_ld_b128(l,      kg);
      async_ld_b128(l + 16, kg + 8);
      async_ld_b128(l + 32, kg + 16);
      async_ld_b128(l + 48, kg + 24);

      // V tile: transpose through VGPRs -> Vts[e][kk].
      const bf16* Vg = Vb + (((size_t)bh * SEQ) + kb * 64) * HD;
      const v8bf* sv = (const v8bf*)(Vg + row * HD + c0);
      v8bf vv0 = sv[0], vv1 = sv[1], vv2 = sv[2], vv3 = sv[3];
#pragma unroll
      for (int i = 0; i < 8; ++i) {
        Vts[(c0 +  0 + i) * 64 + row] = vv0[i];
        Vts[(c0 +  8 + i) * 64 + row] = vv1[i];
        Vts[(c0 + 16 + i) * 64 + row] = vv2[i];
        Vts[(c0 + 24 + i) * 64 + row] = vv3[i];
      }
    }
    wait_async0();     // K (and, on kb==0, Q) copies landed in LDS
    __syncthreads();   // + DScnt for the V transpose, all waves visible

    // ---- S = Q K^T  (16 rows x 64 cols per wave) ----
    v8f s[4] = {};
#pragma unroll
    for (int ks = 0; ks < HD; ks += 32) {
      v16bf a = load_frag_a(Qs, 64, wave * 16, ks);
#pragma unroll
      for (int j = 0; j < 4; ++j) {
        v16bf b = load_frag_b(Ks, 64, ks, j * 16);
        s[j] = wmma_bf16(a, b, s[j]);
      }
    }

    // ---- online softmax (row = (vgpr r, half); reductions stay in half) ----
#pragma unroll
    for (int r = 0; r < 8; ++r) {
      float mx = fmaxf(fmaxf(s[0][r], s[1][r]), fmaxf(s[2][r], s[3][r]));
#pragma unroll
      for (int t = 8; t >= 1; t >>= 1) mx = fmaxf(mx, __shfl_xor(mx, t, 32));
      float mn    = fmaxf(mst[r], mx);
      float alpha = __expf(mst[r] - mn);
      float rs = 0.0f;
#pragma unroll
      for (int j = 0; j < 4; ++j) {
        float p = __expf(s[j][r] - mn);
        s[j][r] = p;
        rs += p;
      }
#pragma unroll
      for (int t = 8; t >= 1; t >>= 1) rs += __shfl_xor(rs, t, 32);
      lst[r] = lst[r] * alpha + rs;
      mst[r] = mn;
      o[0][r] *= alpha; o[1][r] *= alpha; o[2][r] *= alpha; o[3][r] *= alpha;
    }

    // ---- P (C-layout fp32) -> Ps (A-layout bf16) via LDS, wave-private ----
    int prow = wave * 16 + half * 8;
#pragma unroll
    for (int j = 0; j < 4; ++j)
#pragma unroll
      for (int r = 0; r < 8; ++r)
        Ps[(prow + r) * 64 + j * 16 + nc] = f2bf(s[j][r]);
    asm volatile("s_wait_dscnt 0" ::: "memory");   // intra-wave LDS RAW

    // ---- O += P V ----
#pragma unroll
    for (int ks = 0; ks < 64; ks += 32) {
      v16bf a = load_frag_a(Ps, 64, wave * 16, ks);
#pragma unroll
      for (int j = 0; j < 4; ++j) {
        v16bf b = load_frag_b(Vts, 64, ks, j * 16);
        o[j] = wmma_bf16(a, b, o[j]);
      }
    }
  }

  // ---- epilogue: O /= l, store bf16 context in [B,N,C] for the proj GEMM ----
  const int b = bh / HEADS;
  const int h = bh - b * HEADS;
#pragma unroll
  for (int j = 0; j < 4; ++j)
#pragma unroll
    for (int r = 0; r < 8; ++r) {
      int q = qb * 64 + wave * 16 + half * 8 + r;
      int e = j * 16 + nc;
      size_t m = (size_t)b * SEQ + q;
      ctx[m * EMBED + h * HD + e] = f2bf(o[j][r] / lst[r]);
    }
}

// ---------------------------------------------------------------------------
// Host launcher
// ---------------------------------------------------------------------------
extern "C" void kernel_launch(void* const* d_in, const int* in_sizes, int n_in,
                              void* d_out, int out_size, void* d_ws, size_t ws_size,
                              hipStream_t stream) {
  const float* x      = (const float*)d_in[0];
  const float* qkv_w  = (const float*)d_in[1];
  const float* qkv_b  = (const float*)d_in[2];
  const float* proj_w = (const float*)d_in[3];
  const float* proj_b = (const float*)d_in[4];
  float* out = (float*)d_out;

  char* ws = (char*)d_ws;
  size_t off = 0;
  bf16* xb    = (bf16*)(ws + off); off += (size_t)TOKENS * EMBED * sizeof(bf16);
  bf16* wqkvb = (bf16*)(ws + off); off += (size_t)QKV_DIM * EMBED * sizeof(bf16);
  bf16* wprjb = (bf16*)(ws + off); off += (size_t)EMBED * EMBED * sizeof(bf16);
  bf16* Qb    = (bf16*)(ws + off); off += (size_t)BATCH * HEADS * SEQ * HD * sizeof(bf16);
  bf16* Kb    = (bf16*)(ws + off); off += (size_t)BATCH * HEADS * SEQ * HD * sizeof(bf16);
  bf16* Vb    = (bf16*)(ws + off); off += (size_t)BATCH * HEADS * SEQ * HD * sizeof(bf16);
  bf16* ctxb  = (bf16*)(ws + off); off += (size_t)TOKENS * EMBED * sizeof(bf16);

  // 1) repack fp32 -> bf16
  int nx = TOKENS * EMBED, nw1 = QKV_DIM * EMBED, nw2 = EMBED * EMBED;
  cvt_f32_bf16<<<(nx  + 255) / 256, 256, 0, stream>>>(x,      xb,    nx);
  cvt_f32_bf16<<<(nw1 + 255) / 256, 256, 0, stream>>>(qkv_w,  wqkvb, nw1);
  cvt_f32_bf16<<<(nw2 + 255) / 256, 256, 0, stream>>>(proj_w, wprjb, nw2);

  // 2) QKV projection + bias + head split (Q pre-scaled)
  gemm_bf16_kernel<<<dim3(TOKENS / 64, QKV_DIM / 64), 128, 0, stream>>>(
      xb, wqkvb, qkv_b, nullptr, Qb, Kb, Vb, TOKENS, EMBED, QKV_DIM, 1);

  // 3) flash attention per (b,h,query-tile)
  attn_kernel<<<BATCH * HEADS * (SEQ / 64), 128, 0, stream>>>(Qb, Kb, Vb, ctxb);

  // 4) output projection + bias -> fp32 out
  gemm_bf16_kernel<<<dim3(TOKENS / 64, EMBED / 64), 128, 0, stream>>>(
      ctxb, wprjb, proj_b, out, nullptr, nullptr, nullptr, TOKENS, EMBED, EMBED, 0);
}
